// HetGNN_27195732918819
// MI455X (gfx1250) — matrix operand order repaired and checked by reading, not compiled, not run on hardware
//
#include <hip/hip_runtime.h>

typedef __attribute__((ext_vector_type(16))) _Float16 v16h;
typedef __attribute__((ext_vector_type(8)))  _Float16 v8h;
typedef __attribute__((ext_vector_type(2)))  _Float16 h2;
typedef __attribute__((ext_vector_type(8)))  float    v8f;

#define NG 16384
#define NPAIR (NG / 2)
#define STR 72            // activation row stride in halves (144B, keeps b128 alignment)
#define NWAVES 4
// per-wave scratch layout (halves): actA(8x72) actE(16x72) pa(8x72) pe(16x72)
//                                   M1(16x72) agg(16x72) T(16x72) T2(16x72)
#define O_ACTA 0
#define O_ACTE 576
#define O_PA   1728
#define O_PE   2304
#define O_M1   3456
#define O_AGG  4608
#define O_T    5760
#define O_T2   6912
#define PERWAVE 8064

struct Params { const float* in[26]; float* out; };

#define WMMA(a, b, c) \
  __builtin_amdgcn_wmma_f32_16x16x32_f16(false, (a), false, (b), (short)0, (c), false, false)

// ---- WMMA operand loaders (CDNA5 16-bit layouts, ISA 7.12.2) ----------------
// A (16x32, MxK): lane&15 = row M, lane>>4 picks K-halves {0..7,16..23} vs {8..15,24..31}
__device__ __forceinline__ v16h loadA16(const _Float16* base, int stride,
                                        int rsh, int rmask, int k0, int lane) {
  int m = lane & 15, kh = lane >> 4;
  int row = (m >> rsh) & rmask;
  const _Float16* p = base + row * stride + k0 + kh * 8;
  v8h lo = *(const v8h*)p;          // K rel 0..7   (or 8..15)
  v8h hi = *(const v8h*)(p + 16);   // K rel 16..23 (or 24..31)
  v16h r;
#pragma unroll
  for (int i = 0; i < 8; ++i) { r[i] = lo[i]; r[i + 8] = hi[i]; }
  return r;
}

// B (32x16, KxN) from transposed weight unit Wt[n][k] (64x64, stride 64):
// lane&15 = col N, lane>>4 picks K-halves {0..7,16..23} vs {8..15,24..31} (matches A)
__device__ __forceinline__ v16h loadB16(const _Float16* wt, int n0, int k0, int lane) {
  int n = lane & 15, kq = lane >> 4;
  const _Float16* p = wt + (n0 + n) * 64 + k0 + kq * 8;
  v8h lo = *(const v8h*)p;
  v8h hi = *(const v8h*)(p + 16);
  v16h r;
#pragma unroll
  for (int i = 0; i < 8; ++i) { r[i] = lo[i]; r[i + 8] = hi[i]; }
  return r;
}

// out[16x64] = act( A0@W0 [+ A1@W1] + bias ); all 4 N-tiles accumulated in
// registers before any store so B ds_loads can batch/overlap with WMMA issue.
template <bool DUAL, bool HASBIAS, bool RELU, int OUTROWS>
__device__ __forceinline__ void tileLinear(
    const _Float16* a0, int s0, int rsh0, int rm0,
    const _Float16* a1, int s1, int rsh1, int rm1,
    const _Float16* w0, const _Float16* w1,
    const float* bias,
    _Float16* out, int outStride, int lane) {
  v16h A00 = loadA16(a0, s0, rsh0, rm0, 0, lane);
  v16h A01 = loadA16(a0, s0, rsh0, rm0, 32, lane);
  v16h A10 = {}, A11 = {};
  if (DUAL) { A10 = loadA16(a1, s1, rsh1, rm1, 0, lane);
              A11 = loadA16(a1, s1, rsh1, rm1, 32, lane); }
  const int col = lane & 15, half = lane >> 4;
  v8f acc[4];
#pragma unroll
  for (int nt = 0; nt < 4; ++nt) {
    float bv = HASBIAS ? bias[nt * 16 + col] : 0.0f;
    acc[nt] = {bv, bv, bv, bv, bv, bv, bv, bv};
  }
#pragma unroll
  for (int nt = 0; nt < 4; ++nt) {
    acc[nt] = WMMA(A00, loadB16(w0, nt * 16, 0,  lane), acc[nt]);
    acc[nt] = WMMA(A01, loadB16(w0, nt * 16, 32, lane), acc[nt]);
    if (DUAL) {
      acc[nt] = WMMA(A10, loadB16(w1, nt * 16, 0,  lane), acc[nt]);
      acc[nt] = WMMA(A11, loadB16(w1, nt * 16, 32, lane), acc[nt]);
    }
  }
#pragma unroll
  for (int nt = 0; nt < 4; ++nt) {
#pragma unroll
    for (int i = 0; i < 8; ++i) {
      float v = acc[nt][i];
      if (RELU) v = fmaxf(v, 0.0f);
      int m = i + half * 8;                  // D layout: vgpr i, lane>=16 -> M=i+8
      if (OUTROWS == 16 || m < OUTROWS)
        out[m * outStride + nt * 16 + col] = (_Float16)v;
    }
  }
}

// Final M2 layer for one tile (g-local, bp-pair): relu(A@W + b), then fold the
// bp!=b masked max into agg via shfl_xor(16) pair-max + LDS read-modify-write.
__device__ __forceinline__ void tileM2Final(
    const _Float16* a, int s, const _Float16* wt, const float* bias,
    _Float16* agg, int aggStride, int gl, int bph, int lane) {
  v16h A0 = loadA16(a, s, 0, 15, 0, lane);
  v16h A1 = loadA16(a, s, 0, 15, 32, lane);
  const int col = lane & 15, kq = lane >> 4;
  const int bp = bph * 2 + kq;   // tile rows m<8 -> bp_base, m>=8 -> bp_base+1
  v8f acc[4];
#pragma unroll
  for (int nt = 0; nt < 4; ++nt) {
    float bv = bias[nt * 16 + col];
    acc[nt] = {bv, bv, bv, bv, bv, bv, bv, bv};
  }
#pragma unroll
  for (int nt = 0; nt < 4; ++nt) {
    acc[nt] = WMMA(A0, loadB16(wt, nt * 16, 0,  lane), acc[nt]);
    acc[nt] = WMMA(A1, loadB16(wt, nt * 16, 32, lane), acc[nt]);
  }
#pragma unroll
  for (int nt = 0; nt < 4; ++nt) {
#pragma unroll
    for (int i = 0; i < 8; ++i) {
      int b = (i >> 1) & 3, kk = i & 1;
      float v = fmaxf(acc[nt][i], 0.0f);
      if (bp == b) v = -1e30f;               // exclude diagonal bp==b
      float o = __shfl_xor(v, 16);
      v = fmaxf(v, o);                        // max over the bp pair
      if (kq == 0) {                          // single writer per agg entry
        _Float16* pa = agg + (gl * 8 + b * 2 + kk) * aggStride + nt * 16 + col;
        float old = (float)(*pa);
        *pa = (_Float16)fmaxf(old, v);
      }
    }
  }
}

__global__ __launch_bounds__(128, 1) void hetgnn_kernel(Params P) {
  __shared__ __align__(16) _Float16 sW[12 * 4096];   // 12 transposed 64x64 units
  __shared__ float sBias[9 * 64];
  __shared__ float sPre[512];
  __shared__ __align__(16) _Float16 sAct[NWAVES * PERWAVE];

  const int tid = threadIdx.x;
  // ---- one-time: fp32 -> f16 transposed weights into LDS --------------------
  // units: 0 w5_1a 1 w5_1b 2 w5_2 3 w5_3 4 w6_1a 5 w6_1b 6 w6_2 7 w6_3
  //        8 w7_1a 9 w7_1b 10 w7_2 11 w7_3
  const int srcIdx[12] = {6, 6, 8, 10, 12, 12, 14, 16, 18, 18, 20, 22};
  const int rowOff[12] = {0, 64, 0, 0, 0, 64, 0, 0, 0, 64, 0, 0};
#pragma unroll
  for (int u = 0; u < 12; ++u) {
    const float* src = P.in[srcIdx[u]] + rowOff[u] * 64;
    _Float16* dst = &sW[u * 4096];
#pragma unroll 1
    for (int idx = tid; idx < 4096; idx += 128) {
      int n = idx >> 6, kk = idx & 63;
      dst[n * 64 + kk] = (_Float16)src[kk * 64 + n];   // Wt[n][k] = W[k][n]
    }
  }
  const int bidx[9] = {7, 9, 11, 13, 15, 17, 19, 21, 23};
#pragma unroll
  for (int u = 0; u < 9; ++u)
    for (int idx = tid; idx < 64; idx += 128) sBias[u * 64 + idx] = P.in[bidx[u]][idx];
  for (int idx = tid; idx < 64; idx += 128) {
    sPre[idx]        = P.in[2][idx];          // w_pre_ap [1,64]
    sPre[64 + idx]   = P.in[3][idx];          // b_pre_ap
    sPre[128 + idx]  = P.in[4][idx];          // w_pre_edge row 0
    sPre[192 + idx]  = P.in[4][64 + idx];     // w_pre_edge row 1
    sPre[256 + idx]  = P.in[5][idx];          // b_pre_edge
    sPre[320 + 2*idx]     = P.in[24][2*idx];      // w_post [64,2]
    sPre[320 + 2*idx + 1] = P.in[24][2*idx + 1];
  }
  if (tid < 2) sPre[448 + tid] = P.in[25][tid];  // b_post
  __syncthreads();

  const int lane = tid & 31, wave = tid >> 5;
  _Float16* wb   = sAct + wave * PERWAVE;
  _Float16* actA = wb + O_ACTA;  _Float16* actE = wb + O_ACTE;
  _Float16* bPA  = wb + O_PA;    _Float16* bPE  = wb + O_PE;
  _Float16* bM1  = wb + O_M1;    _Float16* bAgg = wb + O_AGG;
  _Float16* bT   = wb + O_T;     _Float16* bT2  = wb + O_T2;
  const _Float16* U = sW;
  const float* ap = P.in[0];
  const float* ef = P.in[1];
  const int stridePairs = gridDim.x * NWAVES;

#pragma unroll 1
  for (int p = blockIdx.x * NWAVES + wave; p < NPAIR; p += stridePairs) {
    const int g0 = 2 * p;
    // prefetch next pair's edge features
    if (p + stridePairs < NPAIR)
      __builtin_prefetch(ef + (size_t)(p + stridePairs) * 32, 0, 1);

    // ---- PreLayer: a = relu(x*w+b) [8x64], e = relu(f@w+b) [16x64] ---------
#pragma unroll 1
    for (int idx = lane; idx < 8 * 32; idx += 32) {     // 2 cols / iter
      int r = idx >> 5, n2 = (idx & 31) * 2;
      int g = g0 + (r >> 2), b = r & 3;
      float x = ap[g * 4 + b];
      h2 o;
      o[0] = (_Float16)fmaxf(x * sPre[n2]     + sPre[64 + n2],     0.0f);
      o[1] = (_Float16)fmaxf(x * sPre[n2 + 1] + sPre[64 + n2 + 1], 0.0f);
      *(h2*)&actA[r * STR + n2] = o;
    }
#pragma unroll 1
    for (int idx = lane; idx < 16 * 32; idx += 32) {
      int r = idx >> 5, n2 = (idx & 31) * 2;
      int g = g0 + (r >> 3), b = (r >> 1) & 3, k = r & 1;
      const float* e2 = ef + (((g * 4 + b) * 2 + k) * 2);
      h2 o;
      o[0] = (_Float16)fmaxf(e2[0] * sPre[128 + n2]     + e2[1] * sPre[192 + n2]     + sPre[256 + n2],     0.0f);
      o[1] = (_Float16)fmaxf(e2[0] * sPre[128 + n2 + 1] + e2[1] * sPre[192 + n2 + 1] + sPre[256 + n2 + 1], 0.0f);
      *(h2*)&actE[r * STR + n2] = o;
    }

    // ---- two UpdateLayers (shared weights) ---------------------------------
#pragma unroll 1
    for (int it = 0; it < 2; ++it) {
      // M1 = mlp5(cat(a_bcast, e)); rows r=(gl*8+b*2+k), a-row = r>>1
      tileLinear<true, true, true, 16>(actA, STR, 1, 7, actE, STR, 0, 15,
                 &U[0 * 4096], &U[1 * 4096], &sBias[0 * 64], bT, STR, lane);
      tileLinear<false, true, true, 16>(bT, STR, 0, 15, nullptr, 0, 0, 0,
                 &U[2 * 4096], nullptr, &sBias[1 * 64], bT2, STR, lane);
      tileLinear<false, true, true, 16>(bT2, STR, 0, 15, nullptr, 0, 0, 0,
                 &U[3 * 4096], nullptr, &sBias[2 * 64], bM1, STR, lane);
      // pa = a@w6_1a + b6_1 (no relu), pe = e@w6_1b (no bias/relu)
      tileLinear<false, true, false, 8>(actA, STR, 0, 7, nullptr, 0, 0, 0,
                 &U[4 * 4096], nullptr, &sBias[3 * 64], bPA, STR, lane);
      tileLinear<false, false, false, 16>(actE, STR, 0, 15, nullptr, 0, 0, 0,
                 &U[5 * 4096], nullptr, nullptr, bPE, STR, lane);
      // agg init = m1max over j!=k  ==  M1 row with k flipped (K=2)
#pragma unroll 1
      for (int idx = lane; idx < 16 * 32; idx += 32) {
        int r = idx >> 5, n2 = (idx & 31) * 2;
        *(h2*)&bAgg[r * STR + n2] = *(const h2*)&bM1[(r ^ 1) * STR + n2];
      }
      // M2: 4 tiles of 16 rows = (gl, bp-pair); h = relu(pa + pe (+b6_1))
#pragma unroll 1
      for (int t = 0; t < 4; ++t) {
        int gl = t >> 1, bph = t & 1;
#pragma unroll 1
        for (int idx = lane; idx < 16 * 32; idx += 32) {
          int m = idx >> 5, n2 = (idx & 31) * 2;
          int bp = bph * 2 + (m >> 3), b = (m >> 1) & 3, kk = m & 1;
          h2 xa = *(const h2*)&bPA[(gl * 4 + b) * STR + n2];
          h2 xe = *(const h2*)&bPE[(gl * 8 + bp * 2 + kk) * STR + n2];
          h2 o;
          o[0] = (_Float16)fmaxf((float)xa[0] + (float)xe[0], 0.0f);
          o[1] = (_Float16)fmaxf((float)xa[1] + (float)xe[1], 0.0f);
          *(h2*)&bT[m * STR + n2] = o;
        }
        tileLinear<false, true, true, 16>(bT, STR, 0, 15, nullptr, 0, 0, 0,
                   &U[6 * 4096], nullptr, &sBias[4 * 64], bT2, STR, lane);
        tileM2Final(bT2, STR, &U[7 * 4096], &sBias[5 * 64], bAgg, STR, gl, bph, lane);
      }
      // e_new = mlp7(cat(agg, e))
      tileLinear<true, true, true, 16>(bAgg, STR, 0, 15, actE, STR, 0, 15,
                 &U[8 * 4096], &U[9 * 4096], &sBias[6 * 64], bT, STR, lane);
      tileLinear<false, true, true, 16>(bT, STR, 0, 15, nullptr, 0, 0, 0,
                 &U[10 * 4096], nullptr, &sBias[7 * 64], bT2, STR, lane);
      tileLinear<false, true, true, 16>(bT2, STR, 0, 15, nullptr, 0, 0, 0,
                 &U[11 * 4096], nullptr, &sBias[8 * 64], actE, STR, lane);
    }

    // ---- post: out = e @ w_post + b_post, L2-normalize over K per (g,b) ----
    {
      int row = lane >> 1, ch = lane & 1;     // 16 rows x 2 complex components
      float acc = sPre[448 + ch];
#pragma unroll 1
      for (int n = 0; n < 64; ++n)
        acc += (float)actE[row * STR + n] * sPre[320 + 2 * n + ch];
      float sq = acc * acc;
      sq += __shfl_xor(sq, 1);                // sum r^2+im^2 over k and component
      sq += __shfl_xor(sq, 2);
      P.out[(size_t)g0 * 16 + lane] = acc * rsqrtf(sq);
    }
  }
}

extern "C" void kernel_launch(void* const* d_in, const int* in_sizes, int n_in,
                              void* d_out, int out_size, void* d_ws, size_t ws_size,
                              hipStream_t stream) {
  (void)in_sizes; (void)n_in; (void)out_size; (void)d_ws; (void)ws_size;
  Params P;
  for (int i = 0; i < 26; ++i) P.in[i] = (const float*)d_in[i];
  P.out = (float*)d_out;
  hipLaunchKernelGGL(hetgnn_kernel, dim3(512), dim3(128), 0, stream, P);
}
